// SpatialTransformer_24481313587910
// MI455X (gfx1250) — matrix operand order repaired and checked by reading, not compiled, not run on hardware
//
#include <hip/hip_runtime.h>
#include <hip/hip_bf16.h>

// ---------------- problem constants ----------------
#define NT    4096          // tokens = 16*16*16
#define CIN   256
#define NGRP  32
#define INNER 512
#define HEADS 8
#define HD    64
#define FFD   2048

typedef __attribute__((ext_vector_type(16))) __bf16 v16bf;
typedef __attribute__((ext_vector_type(8)))  float  v8f;

union Frag { uint4 u[2]; v16bf v; };

// fp32 -> bf16 (round to nearest even)
__device__ __forceinline__ unsigned short f2bf(float f) {
    unsigned int u = __float_as_uint(f);
    unsigned int r = u + 0x7FFFu + ((u >> 16) & 1u);
    return (unsigned short)(r >> 16);
}

__device__ __forceinline__ v8f wmma_bf16(v16bf a, v16bf b, v8f c) {
    return __builtin_amdgcn_wmma_f32_16x16x32_bf16(false, a, false, b, (short)0, c, false, false);
}

// A-fragment: 16x32 bf16 tile, rows m0..m0+15, cols k0..k0+31 of row-major [*, lda]
// lane<16 : row=lane,   K = {k0+0..7, k0+16..23}
// lane>=16: row=lane-16,K = {k0+8..15,k0+24..31}
__device__ __forceinline__ v16bf load_fragA(const unsigned short* base, int lda,
                                            int m0, int k0, int lane) {
    int hf = lane >> 4, r = lane & 15;
    const unsigned short* p = base + (size_t)(m0 + r) * lda + k0 + hf * 8;
    Frag f;
    f.u[0] = *(const uint4*)(p);
    f.u[1] = *(const uint4*)(p + 16);
    return f.v;
}

// B-fragment: 32x16, B[k][n] = W[n0+n][k0+k] from row-major W[*, ldw]
// lane<16 : col=lane,   K = {k0+0..7, k0+16..23}
// lane>=16: col=lane-16,K = {k0+8..15,k0+24..31}
__device__ __forceinline__ v16bf load_fragB(const unsigned short* base, int ldw,
                                            int n0, int k0, int lane) {
    int hf = lane >> 4, c = lane & 15;
    const unsigned short* p = base + (size_t)(n0 + c) * ldw + k0 + hf * 8;
    Frag f;
    f.u[0] = *(const uint4*)(p);
    f.u[1] = *(const uint4*)(p + 16);
    return f.v;
}

// ---- CDNA5 async copy: 16B global -> LDS, tracked by ASYNCcnt ----
__device__ __forceinline__ void async_copy16(const unsigned short* gp, unsigned short* lp) {
    unsigned loff = (unsigned)(__UINTPTR_TYPE__)(__attribute__((address_space(3))) unsigned short*)lp;
    asm volatile("global_load_async_to_lds_b128 %0, %1, off"
                 :: "v"(loff), "v"(gp) : "memory");
}
__device__ __forceinline__ void wait_async0() {
    asm volatile("s_wait_asynccnt 0x0" ::: "memory");
}

// ---------------- fp32 -> bf16 convert ----------------
__global__ void k_f2bf(const float* __restrict__ src, unsigned short* __restrict__ dst, int n) {
    int i = blockIdx.x * blockDim.x + threadIdx.x;
    if (i < n) dst[i] = f2bf(src[i]);
}

// ---------------- GroupNorm stats: 32 groups, 8 ch * 4096 each ----------------
__global__ __launch_bounds__(256) void k_gn_stats(const float* __restrict__ x,
                                                  float* __restrict__ stats) {
    int g = blockIdx.x, tid = threadIdx.x;
    float s = 0.f, ss = 0.f;
    for (int idx = tid; idx < 8 * NT; idx += 256) {
        int c = g * 8 + (idx >> 12);
        int n = idx & (NT - 1);
        float v = x[(size_t)c * NT + n];
        s += v; ss += v * v;
    }
    __shared__ float rs[256], rq[256];
    rs[tid] = s; rq[tid] = ss;
    __syncthreads();
    for (int o = 128; o > 0; o >>= 1) {
        if (tid < o) { rs[tid] += rs[tid + o]; rq[tid] += rq[tid + o]; }
        __syncthreads();
    }
    if (tid == 0) {
        float mean = rs[0] / (8.f * NT);
        float var  = rq[0] / (8.f * NT) - mean * mean;
        stats[2 * g]     = mean;
        stats[2 * g + 1] = rsqrtf(var + 1e-5f);
    }
}

// normalize + affine + transpose (C,N) -> bf16 (N,C)
__global__ void k_gn_apply(const float* __restrict__ x, const float* __restrict__ stats,
                           const float* __restrict__ gg, const float* __restrict__ gb,
                           unsigned short* __restrict__ h) {
    int e = blockIdx.x * blockDim.x + threadIdx.x;  // e = n*C + c
    if (e >= NT * CIN) return;
    int c = e & (CIN - 1);
    int n = e >> 8;
    int g = c >> 3;
    float v = (x[(size_t)c * NT + n] - stats[2 * g]) * stats[2 * g + 1] * gg[c] + gb[c];
    h[e] = f2bf(v);
}

// ---------------- WMMA GEMM with async-LDS double buffering ----------------
// Y = act(A(M,K) * W(OUT,K)^T + bias)
// block 256 = 8 waves arranged 4(m) x 2(n); block tile 64 rows x 128 cols
// grid = (OUT/128, M/64). LDS: A 64x32, B 128x32, double buffered (24KB).
// fmode: 0 row-major f32 [M,OUT]; 1 transposed [OUT,NT] + resid
// bmode: 0 row-major bf16; 1 head-major [H][NT][64]; 2 head-T [H][64][NT]
__global__ __launch_bounds__(256) void k_gemm(const unsigned short* __restrict__ A,
                                              const unsigned short* __restrict__ W,
                                              const float* __restrict__ bias,
                                              int K, int OUT,
                                              float* outF, int fmode,
                                              const float* __restrict__ resid,
                                              unsigned short* outB, int bmode, int act) {
    int tid  = threadIdx.x;
    int lane = tid & 31;
    int wv   = tid >> 5;
    int mw   = wv & 3;          // m sub-tile within block
    int nw   = wv >> 2;         // n sub-tile within block
    int mb   = blockIdx.y * 64; // block row base
    int nb   = blockIdx.x * 128;// block col base

    __shared__ __align__(16) unsigned short sA[2][64 * 32];
    __shared__ __align__(16) unsigned short sB[2][128 * 32];

    // stage helpers: tile rows x 32 bf16 -> LDS row-major (stride 32)
    auto stageA = [&](int buf, int k0) {
        int c = tid;                         // 64 rows * 4 chunks = 256
        int r = c >> 2, q = c & 3;
        async_copy16(A + (size_t)(mb + r) * K + k0 + q * 8, &sA[buf][r * 32 + q * 8]);
    };
    auto stageB = [&](int buf, int k0) {
#pragma unroll
        for (int it = 0; it < 2; ++it) {     // 128 rows * 4 chunks = 512
            int c = tid + it * 256;
            int r = c >> 2, q = c & 3;
            async_copy16(W + (size_t)(nb + r) * K + k0 + q * 8, &sB[buf][r * 32 + q * 8]);
        }
    };

    v8f acc[4];
#pragma unroll
    for (int t = 0; t < 4; ++t)
#pragma unroll
        for (int i = 0; i < 8; ++i) acc[t][i] = 0.f;

    stageA(0, 0);
    stageB(0, 0);
    wait_async0();
    __syncthreads();

    for (int k0 = 0; k0 < K; k0 += 32) {
        int cur = (k0 >> 5) & 1;
        if (k0 + 32 < K) { stageA(cur ^ 1, k0 + 32); stageB(cur ^ 1, k0 + 32); }

        v16bf av = load_fragA(sA[cur], 32, mw * 16, 0, lane);
#pragma unroll
        for (int t = 0; t < 4; ++t) {
            v16bf bv = load_fragB(sB[cur], 32, nw * 64 + t * 16, 0, lane);
            acc[t] = wmma_bf16(av, bv, acc[t]);
        }
        wait_async0();
        __syncthreads();
    }

    int cl = lane & 15, hf = lane >> 4;
    int m0 = mb + mw * 16;
    int n0 = nb + nw * 64;
#pragma unroll
    for (int t = 0; t < 4; ++t) {
        int col = n0 + t * 16 + cl;
        float bia = bias ? bias[col] : 0.f;
#pragma unroll
        for (int i = 0; i < 8; ++i) {
            int row = m0 + i + 8 * hf;
            float v = acc[t][i] + bia;
            if (act) v = 0.5f * v * (1.f + erff(v * 0.70710678f));  // exact GELU
            if (outF) {
                if (fmode == 0) outF[(size_t)row * OUT + col] = v;
                else            outF[(size_t)col * NT + row] = v + resid[(size_t)col * NT + row];
            }
            if (outB) {
                size_t idx;
                if (bmode == 0)      idx = (size_t)row * OUT + col;
                else if (bmode == 1) idx = (size_t)(col >> 6) * (NT * 64) + (size_t)row * 64 + (col & 63);
                else                 idx = (size_t)(col >> 6) * (NT * 64) + (size_t)(col & 63) * NT + row;
                outB[idx] = f2bf(v);
            }
        }
    }
}

// ---------------- flash attention: 1 wave per (head, 16-query tile) ----------------
__device__ __forceinline__ float rmax16(float x) {
#pragma unroll
    for (int m = 1; m < 16; m <<= 1) x = fmaxf(x, __shfl_xor(x, m, 32));
    return x;
}
__device__ __forceinline__ float rsum16(float x) {
#pragma unroll
    for (int m = 1; m < 16; m <<= 1) x += __shfl_xor(x, m, 32);
    return x;
}

__global__ __launch_bounds__(32) void k_attn(const unsigned short* __restrict__ Q,
                                             const unsigned short* __restrict__ Km,
                                             const unsigned short* __restrict__ Vt,
                                             float* __restrict__ O) {
    int h = blockIdx.x;
    int q0 = blockIdx.y * 16;
    int lane = threadIdx.x;
    int cl = lane & 15, hf = lane >> 4;

    const unsigned short* qh = Q  + (size_t)h * NT * HD;   // [NT][64]
    const unsigned short* kh = Km + (size_t)h * NT * HD;   // [NT][64]
    const unsigned short* vh = Vt + (size_t)h * HD * NT;   // [64][NT]

    v16bf a0 = load_fragA(qh, HD, q0, 0,  lane);
    v16bf a1 = load_fragA(qh, HD, q0, 32, lane);

    v8f oacc[4];
    float mrow[8], lrow[8];
#pragma unroll
    for (int t = 0; t < 4; ++t)
#pragma unroll
        for (int i = 0; i < 8; ++i) oacc[t][i] = 0.f;
#pragma unroll
    for (int i = 0; i < 8; ++i) { mrow[i] = -3.0e38f; lrow[i] = 0.f; }

    __shared__ __align__(16) unsigned short ldsP[16 * 32];

    for (int j0 = 0; j0 < NT; j0 += 32) {
        v8f s0, s1;
#pragma unroll
        for (int i = 0; i < 8; ++i) { s0[i] = 0.f; s1[i] = 0.f; }
        s0 = wmma_bf16(a0, load_fragB(kh, HD, j0,      0,  lane), s0);
        s0 = wmma_bf16(a1, load_fragB(kh, HD, j0,      32, lane), s0);
        s1 = wmma_bf16(a0, load_fragB(kh, HD, j0 + 16, 0,  lane), s1);
        s1 = wmma_bf16(a1, load_fragB(kh, HD, j0 + 16, 32, lane), s1);

        // online softmax (rows striped: element i -> row i + 8*hf; cols across 16 lanes)
#pragma unroll
        for (int i = 0; i < 8; ++i) {
            float x0 = s0[i] * 0.125f;            // HD^-0.5
            float x1 = s1[i] * 0.125f;
            float tmax = rmax16(fmaxf(x0, x1));
            float mn = fmaxf(mrow[i], tmax);
            float cor = __expf(mrow[i] - mn);
            float p0 = __expf(x0 - mn);
            float p1 = __expf(x1 - mn);
            lrow[i] = lrow[i] * cor + rsum16(p0 + p1);
            mrow[i] = mn;
#pragma unroll
            for (int t = 0; t < 4; ++t) oacc[t][i] *= cor;
            int r = i + 8 * hf;
            ldsP[r * 32 + cl]      = f2bf(p0);
            ldsP[r * 32 + 16 + cl] = f2bf(p1);
        }
        __syncthreads();

        v16bf pa = load_fragA(ldsP, 32, 0, 0, lane);   // P 16x32 -> A layout
#pragma unroll
        for (int t = 0; t < 4; ++t) {
            v16bf bv = load_fragB(vh, NT, t * 16, j0, lane);  // V^T gives contiguous K
            oacc[t] = wmma_bf16(pa, bv, oacc[t]);
        }
        __syncthreads();
    }

#pragma unroll
    for (int t = 0; t < 4; ++t) {
        int col = h * 64 + t * 16 + cl;
#pragma unroll
        for (int i = 0; i < 8; ++i) {
            int row = q0 + i + 8 * hf;
            float inv = lrow[i] > 0.f ? 1.f / lrow[i] : 0.f;
            O[(size_t)row * INNER + col] = oacc[t][i] * inv;
        }
    }
}

// ---------------- fused residual-add + LayerNorm over 512 ----------------
__global__ __launch_bounds__(256) void k_add_ln(const float* __restrict__ a,
                                                const float* __restrict__ b,
                                                const float* __restrict__ g,
                                                const float* __restrict__ bb,
                                                float* outF, unsigned short* __restrict__ outB) {
    int r = blockIdx.x, tid = threadIdx.x;
    float v0 = a[(size_t)r * INNER + tid]       + b[(size_t)r * INNER + tid];
    float v1 = a[(size_t)r * INNER + 256 + tid] + b[(size_t)r * INNER + 256 + tid];

    __shared__ float red[256];
    __shared__ float smu, srs;
    red[tid] = v0 + v1;
    __syncthreads();
    for (int o = 128; o > 0; o >>= 1) { if (tid < o) red[tid] += red[tid + o]; __syncthreads(); }
    if (tid == 0) smu = red[0] / (float)INNER;
    __syncthreads();
    float mu = smu;
    red[tid] = (v0 - mu) * (v0 - mu) + (v1 - mu) * (v1 - mu);
    __syncthreads();
    for (int o = 128; o > 0; o >>= 1) { if (tid < o) red[tid] += red[tid + o]; __syncthreads(); }
    if (tid == 0) srs = rsqrtf(red[0] / (float)INNER + 1e-5f);
    __syncthreads();
    float rstd = srs;

    float y0 = (v0 - mu) * rstd * g[tid]       + bb[tid];
    float y1 = (v1 - mu) * rstd * g[tid + 256] + bb[tid + 256];
    if (outF) {
        outF[(size_t)r * INNER + tid]       = y0;
        outF[(size_t)r * INNER + 256 + tid] = y1;
    }
    outB[(size_t)r * INNER + tid]       = f2bf(y0);
    outB[(size_t)r * INNER + 256 + tid] = f2bf(y1);
}

// ---------------- host orchestration ----------------
extern "C" void kernel_launch(void* const* d_in, const int* in_sizes, int n_in,
                              void* d_out, int out_size, void* d_ws, size_t ws_size,
                              hipStream_t stream) {
    const float* x     = (const float*)d_in[0];
    const float* gn_g  = (const float*)d_in[1];
    const float* gn_b  = (const float*)d_in[2];
    const float* w_in  = (const float*)d_in[3];
    const float* b_in  = (const float*)d_in[4];
    const float* wq    = (const float*)d_in[5];
    const float* wk    = (const float*)d_in[6];
    const float* wvv   = (const float*)d_in[7];
    const float* ln1g  = (const float*)d_in[8];
    const float* ln1b  = (const float*)d_in[9];
    const float* ffw1  = (const float*)d_in[10];
    const float* ffb1  = (const float*)d_in[11];
    const float* ffw2  = (const float*)d_in[12];
    const float* ffb2  = (const float*)d_in[13];
    const float* ln2g  = (const float*)d_in[14];
    const float* ln2b  = (const float*)d_in[15];
    const float* w_out = (const float*)d_in[16];
    const float* b_out = (const float*)d_in[17];

    char* ws = (char*)d_ws;
    size_t off = 0;
    auto alloc = [&](size_t bytes) -> void* {
        void* p = ws + off;
        off = (off + bytes + 255) & ~(size_t)255;
        return p;
    };

    float* gstats           = (float*)alloc(NGRP * 2 * 4);
    unsigned short* h_bf    = (unsigned short*)alloc((size_t)NT * CIN * 2);
    float* t_f32            = (float*)alloc((size_t)NT * INNER * 4);
    unsigned short* t_bf    = (unsigned short*)alloc((size_t)NT * INNER * 2);
    unsigned short* q_bf    = (unsigned short*)alloc((size_t)NT * INNER * 2);
    unsigned short* kk_bf   = (unsigned short*)alloc((size_t)NT * INNER * 2);
    unsigned short* v_bf    = (unsigned short*)alloc((size_t)NT * INNER * 2);
    float* o_f32            = (float*)alloc((size_t)NT * INNER * 4);
    float* u_f32            = (float*)alloc((size_t)NT * INNER * 4);
    unsigned short* u_bf    = (unsigned short*)alloc((size_t)NT * INNER * 2);
    unsigned short* gelu_bf = (unsigned short*)alloc((size_t)NT * FFD * 2);
    unsigned short* z_bf    = (unsigned short*)alloc((size_t)NT * INNER * 2);
    float* f_f32            = o_f32;  // reuse: attention out dead after ln1

    unsigned short* w_in_bf = (unsigned short*)alloc((size_t)INNER * CIN * 2);
    unsigned short* wq_bf   = (unsigned short*)alloc((size_t)INNER * INNER * 2);
    unsigned short* wk_bf   = (unsigned short*)alloc((size_t)INNER * INNER * 2);
    unsigned short* wv_bf   = (unsigned short*)alloc((size_t)INNER * INNER * 2);
    unsigned short* w1_bf   = (unsigned short*)alloc((size_t)FFD * INNER * 2);
    unsigned short* w2_bf   = (unsigned short*)alloc((size_t)INNER * FFD * 2);
    unsigned short* wo_bf   = (unsigned short*)alloc((size_t)CIN * INNER * 2);

    auto cvt = [&](const float* s, unsigned short* d, int n) {
        k_f2bf<<<(n + 255) / 256, 256, 0, stream>>>(s, d, n);
    };
    cvt(w_in,  w_in_bf, INNER * CIN);
    cvt(wq,    wq_bf,   INNER * INNER);
    cvt(wk,    wk_bf,   INNER * INNER);
    cvt(wvv,   wv_bf,   INNER * INNER);
    cvt(ffw1,  w1_bf,   FFD * INNER);
    cvt(ffw2,  w2_bf,   INNER * FFD);
    cvt(w_out, wo_bf,   CIN * INNER);

    // GroupNorm
    k_gn_stats<<<NGRP, 256, 0, stream>>>(x, gstats);
    k_gn_apply<<<(NT * CIN) / 256, 256, 0, stream>>>(x, gstats, gn_g, gn_b, h_bf);

    // proj_in: t = h @ w_in^T + b_in   (f32 residual + bf16 operand copies)
    k_gemm<<<dim3(INNER / 128, NT / 64), 256, 0, stream>>>(
        h_bf, w_in_bf, b_in, CIN, INNER, t_f32, 0, nullptr, t_bf, 0, 0);

    // q/k/v projections (no bias); head-major for q,k; head-transposed for v
    k_gemm<<<dim3(INNER / 128, NT / 64), 256, 0, stream>>>(
        t_bf, wq_bf, nullptr, INNER, INNER, nullptr, 0, nullptr, q_bf, 1, 0);
    k_gemm<<<dim3(INNER / 128, NT / 64), 256, 0, stream>>>(
        t_bf, wk_bf, nullptr, INNER, INNER, nullptr, 0, nullptr, kk_bf, 1, 0);
    k_gemm<<<dim3(INNER / 128, NT / 64), 256, 0, stream>>>(
        t_bf, wv_bf, nullptr, INNER, INNER, nullptr, 0, nullptr, v_bf, 2, 0);

    // flash attention
    k_attn<<<dim3(HEADS, NT / 16), 32, 0, stream>>>(q_bf, kk_bf, v_bf, o_f32);

    // ln1(o + t)
    k_add_ln<<<NT, 256, 0, stream>>>(o_f32, t_f32, ln1g, ln1b, u_f32, u_bf);

    // FF: GELU(u @ w1^T + b1) @ w2^T + b2
    k_gemm<<<dim3(FFD / 128, NT / 64), 256, 0, stream>>>(
        u_bf, w1_bf, ffb1, INNER, FFD, nullptr, 0, nullptr, gelu_bf, 0, 1);
    k_gemm<<<dim3(INNER / 128, NT / 64), 256, 0, stream>>>(
        gelu_bf, w2_bf, ffb2, FFD, INNER, f_f32, 0, nullptr, nullptr, 0, 0);

    // ln2(ff + u)
    k_add_ln<<<NT, 256, 0, stream>>>(f_f32, u_f32, ln2g, ln2b, nullptr, z_bf);

    // proj_out + residual x, transposed to (C, N) layout
    k_gemm<<<dim3(CIN / 128, NT / 64), 256, 0, stream>>>(
        z_bf, wo_bf, b_out, INNER, CIN, (float*)d_out, 1, x, nullptr, 0, 0);

    (void)in_sizes; (void)n_in; (void)out_size; (void)ws_size;
}
